// HTSAT_Swin_Transformer_28458453303727
// MI455X (gfx1250) — matrix operand order repaired and checked by reading, not compiled, not run on hardware
//
#include <hip/hip_runtime.h>
#include <hip/hip_bf16.h>

// ---------------------------------------------------------------------------
// Swin block on gfx1250: all GEMMs via v_wmma_f32_16x16x32_f16 (wave32 WMMA),
// TDM (tensor_load_to_lds) staging of MLP weight panels into LDS.
// ---------------------------------------------------------------------------

typedef __attribute__((ext_vector_type(16))) _Float16 v16h;
typedef __attribute__((ext_vector_type(8)))  _Float16 v8h;
typedef __attribute__((ext_vector_type(8)))  float    v8f;
typedef unsigned int u32x4 __attribute__((ext_vector_type(4)));
typedef int          i32x4 __attribute__((ext_vector_type(4)));
typedef int          i32x8 __attribute__((ext_vector_type(8)));

#define LL_   4096
#define CC_   96
#define SSZ   4
#define NTOK  64          // tokens per window
#define NWIN  4096        // B * nW
#define NHEAD 4
#define HD    24
#define QSCALE 0.2041241452319315f   // 24^-0.5

__device__ __forceinline__ v8f wmma_f16(v16h a, v16h b, v8f c) {
  return __builtin_amdgcn_wmma_f32_16x16x32_f16(false, a, false, b, (short)0, c, false, false);
}

// ---- fragment loaders, wave32; contiguous k = 16*(L/16) + j, same order in A and B -----------------

// A 16x32 f16: lane L -> row M=L&15; kbase=(L<16)?0:16; half j -> K = kbase + j
__device__ __forceinline__ v16h load_A(const _Float16* __restrict__ p, int ld) {
  int lane = threadIdx.x & 31;
  const _Float16* q = p + (lane & 15) * ld;
  int kb = (lane < 16) ? 0 : 16;
  v16h a;
#pragma unroll
  for (int j = 0; j < 16; ++j) a[j] = q[kb + j];
  return a;
}

__device__ __forceinline__ v16h load_A_k(const _Float16* __restrict__ p, int ld, int kmax) {
  int lane = threadIdx.x & 31;
  const _Float16* q = p + (lane & 15) * ld;
  int kb = (lane < 16) ? 0 : 16;
  v16h a;
#pragma unroll
  for (int j = 0; j < 16; ++j) {
    int k = kb + j;
    a[j] = (k < kmax) ? q[k] : (_Float16)0.0f;
  }
  return a;
}

// B 32x16 f16 from row-major W[n][k] (B(k,n)=W[n][k]); lane -> col N=L&15; kbase=(L<16)?0:16
__device__ __forceinline__ v16h load_Bwt(const _Float16* __restrict__ p, int ld) {
  int lane = threadIdx.x & 31;
  const _Float16* q = p + (lane & 15) * ld + ((lane < 16) ? 0 : 16);
  v16h b;
#pragma unroll
  for (int j = 0; j < 16; ++j) b[j] = q[j];
  return b;
}

__device__ __forceinline__ v16h load_Bwt_k(const _Float16* __restrict__ p, int ld, int kmax) {
  int lane = threadIdx.x & 31;
  int kb = (lane < 16) ? 0 : 16;
  const _Float16* q = p + (lane & 15) * ld;
  v16h b;
#pragma unroll
  for (int j = 0; j < 16; ++j) b[j] = (kb + j < kmax) ? q[kb + j] : (_Float16)0.0f;
  return b;
}

__device__ __forceinline__ v16h load_Bwt_n(const _Float16* __restrict__ p, int ld, int nmax) {
  int lane = threadIdx.x & 31;
  int n = lane & 15;
  const _Float16* q = p + n * ld + ((lane < 16) ? 0 : 16);
  v16h b;
#pragma unroll
  for (int j = 0; j < 16; ++j) b[j] = (n < nmax) ? q[j] : (_Float16)0.0f;
  return b;
}

// ---- reductions (wave32) ----------------------------------------------------

__device__ __forceinline__ float red32(float v) {
#pragma unroll
  for (int m = 16; m >= 1; m >>= 1) v += __shfl_xor(v, m, 32);
  return v;
}

// ---- shift-mask / rel-pos helpers (closed form, no tables) -----------------

__device__ __forceinline__ int mrgn(int p) { return p < 56 ? 0 : (p < 60 ? 1 : 2); }
__device__ __forceinline__ int mval(int w, int tok) {
  int pr = ((w >> 3) << 3) + (tok >> 3);
  int pc = ((w & 7) << 3) + (tok & 7);
  return 3 * mrgn(pr) + mrgn(pc);
}
__device__ __forceinline__ int rpidx(int i, int j) {
  int r0 = (i >> 3) - (j >> 3) + 7;
  int r1 = (i & 7) - (j & 7) + 7;
  return r0 * 15 + r1;        // (2*WS-1) = 15
}

// ---- TDM: DMA a 2D f16 panel (rows x row_elems, row-major, dense) into LDS --
// D# per CDNA5 ISA §8.3/§8.4: group0 {count, lds_addr, global_addr, type=2},
// group1 {data_size=2B, tensor_dim0/1, tile_dim0/1, tensor_dim0_stride}.
__device__ __forceinline__ void tdm_load_2d(const void* gsrc, unsigned lds_byte_off,
                                            unsigned row_elems, unsigned rows) {
  unsigned long long ga = (unsigned long long)(uintptr_t)gsrc;
  u32x4 g0;
  g0[0] = 1u;                                                  // count=1, user mode
  g0[1] = lds_byte_off;                                        // lds_addr
  g0[2] = (unsigned)ga;                                        // global_addr[31:0]
  g0[3] = (unsigned)((ga >> 32) & 0x1FFFFFFull) | (2u << 30);  // addr[56:32] | type=2
  i32x8 g1;
  g1[0] = (int)(1u << 16);                                     // data_size code 1 (2B)
  g1[1] = (int)((row_elems & 0xFFFFu) << 16);                  // tensor_dim0[15:0]
  g1[2] = (int)(((row_elems >> 16) & 0xFFFFu) | ((rows & 0xFFFFu) << 16)); // dim0 hi | dim1 lo
  g1[3] = (int)(((rows >> 16) & 0xFFFFu) | ((row_elems & 0xFFFFu) << 16)); // dim1 hi | tile_dim0
  g1[4] = (int)(rows & 0xFFFFu);                               // tile_dim1 (tile_dim2=0)
  g1[5] = (int)row_elems;                                      // tensor_dim0_stride[31:0]
  g1[6] = 0;
  g1[7] = 0;
  i32x4 gz = {0, 0, 0, 0};
#if defined(__clang_major__) && (__clang_major__ >= 23)
  i32x8 gz8 = {0, 0, 0, 0, 0, 0, 0, 0};
  __builtin_amdgcn_tensor_load_to_lds(g0, g1, gz, gz, gz8, 0);
#else
  __builtin_amdgcn_tensor_load_to_lds(g0, g1, gz, gz, 0);
#endif
}

// ---------------------------------------------------------------------------
__global__ void k_cvt(const float* __restrict__ src, _Float16* __restrict__ dst, int n) {
  int i = blockIdx.x * 256 + threadIdx.x;
  if (i < n) dst[i] = (_Float16)src[i];
}

// ---------------------------------------------------------------------------
// k_ln1: LayerNorm1 + roll(-SS) + window partition -> f16 (one wave per token)
// ---------------------------------------------------------------------------
__global__ void k_ln1(const float* __restrict__ x, const float* __restrict__ g,
                      const float* __restrict__ bta, _Float16* __restrict__ h16) {
  int token = blockIdx.x * 8 + (threadIdx.x >> 5);
  int lane  = threadIdx.x & 31;
  int win = token >> 6, n = token & 63;
  int b = win >> 6, w = win & 63;
  int pr = (((w >> 3) << 3) + (n >> 3) + SSZ) & 63;
  int pc = (((w & 7) << 3) + (n & 7) + SSZ) & 63;
  const float* row = x + (((size_t)b * LL_) + pr * 64 + pc) * CC_;
  float v0 = row[lane], v1 = row[lane + 32], v2 = row[lane + 64];
  float mu = red32(v0 + v1 + v2) * (1.0f / 96.0f);
  float d0 = v0 - mu, d1 = v1 - mu, d2 = v2 - mu;
  float var = red32(d0 * d0 + d1 * d1 + d2 * d2) * (1.0f / 96.0f);
  float rs = rsqrtf(var + 1e-5f);
  _Float16* o = h16 + (size_t)token * CC_;
  o[lane]      = (_Float16)(d0 * rs * g[lane]      + bta[lane]);
  o[lane + 32] = (_Float16)(d1 * rs * g[lane + 32] + bta[lane + 32]);
  o[lane + 64] = (_Float16)(d2 * rs * g[lane + 64] + bta[lane + 64]);
}

__global__ void k_ln2(const float* __restrict__ x, const float* __restrict__ g,
                      const float* __restrict__ bta, _Float16* __restrict__ h16) {
  int token = blockIdx.x * 8 + (threadIdx.x >> 5);
  int lane  = threadIdx.x & 31;
  const float* row = x + (size_t)token * CC_;
  float v0 = row[lane], v1 = row[lane + 32], v2 = row[lane + 64];
  float mu = red32(v0 + v1 + v2) * (1.0f / 96.0f);
  float d0 = v0 - mu, d1 = v1 - mu, d2 = v2 - mu;
  float var = red32(d0 * d0 + d1 * d1 + d2 * d2) * (1.0f / 96.0f);
  float rs = rsqrtf(var + 1e-5f);
  _Float16* o = h16 + (size_t)token * CC_;
  o[lane]      = (_Float16)(d0 * rs * g[lane]      + bta[lane]);
  o[lane + 32] = (_Float16)(d1 * rs * g[lane + 32] + bta[lane + 32]);
  o[lane + 64] = (_Float16)(d2 * rs * g[lane + 64] + bta[lane + 64]);
}

// ---------------------------------------------------------------------------
// k_qkv: per-window GEMM h(64x96) @ qkv_w^T(96x288)+b; q pre-scaled.
// Q,K -> qk16[win][n][192] (row-major); V -> vT[win][96][64] (transposed,
// so the PV B-fragments and this store are both fully coalesced).
// ---------------------------------------------------------------------------
__global__ void k_qkv(const _Float16* __restrict__ h16, const _Float16* __restrict__ wq,
                      const float* __restrict__ qb, _Float16* __restrict__ qk16,
                      _Float16* __restrict__ vT) {
  int win = blockIdx.x;
  int wave = threadIdx.x >> 5, lane = threadIdx.x & 31;
  const _Float16* Ab = h16 + (size_t)win * NTOK * CC_;
  int ncol = lane & 15, gsel = lane >> 4;
  for (int t = wave; t < 72; t += 8) {
    int mt = t / 18, nt = t % 18;
    v8f acc = {};
#pragma unroll
    for (int kt = 0; kt < 3; ++kt) {
      v16h a  = load_A(Ab + mt * 16 * CC_ + kt * 32, CC_);
      v16h bm = load_Bwt(wq + (nt * 16) * CC_ + kt * 32, CC_);
      acc = wmma_f16(a, bm, acc);
    }
    int col = nt * 16 + ncol;
    float bias = qb[col];
    if (nt < 12) {              // Q (scaled) and K, row-major [n][192]
      float scl = (col < 96) ? QSCALE : 1.0f;
      _Float16* op = qk16 + (size_t)win * NTOK * 192 + col;
#pragma unroll
      for (int r = 0; r < 8; ++r) {
        int m = mt * 16 + r + (gsel << 3);
        op[(size_t)m * 192] = (_Float16)((acc[r] + bias) * scl);
      }
    } else {                    // V transposed: vT[win][col-192][m], 16B vector store
      v8h pk;
#pragma unroll
      for (int r = 0; r < 8; ++r) pk[r] = (_Float16)(acc[r] + bias);
      *(v8h*)(vT + (size_t)win * 96 * 64 + (size_t)(col - 192) * 64 + mt * 16 + (gsel << 3)) = pk;
    }
  }
}

// ---------------------------------------------------------------------------
// k_attn: per (window, head): QK^T WMMA -> closed-form bias+mask -> softmax
// (16-lane shfl reductions matching C/D layout) -> D->A transpose through LDS
// (ds_store/ds_load + s_wait_dscnt) -> PV WMMA. 4 waves/block (one head each).
// ---------------------------------------------------------------------------
__global__ void k_attn(const _Float16* __restrict__ qk16, const _Float16* __restrict__ vT,
                       const float* __restrict__ btab, _Float16* __restrict__ ao16) {
  __shared__ _Float16 lds[NHEAD][16 * 64];
  int win  = blockIdx.x;
  int head = threadIdx.x >> 5;
  int lane = threadIdx.x & 31;
  int w = win & 63;
  const _Float16* qp = qk16 + (size_t)win * NTOK * 192 + head * HD;
  const _Float16* kp = qp + 96;
  const _Float16* vp = vT + (size_t)win * 96 * 64 + (size_t)head * HD * 64;

  v16h kf[4];
#pragma unroll
  for (int c = 0; c < 4; ++c) kf[c] = load_Bwt_k(kp + (c * 16) * 192, 192, HD);
  // V fragments from vT: B(k=m, n=d) = vT[d][m] -> wt-form, ld=64, fully coalesced
  v16h vf[2][2];
#pragma unroll
  for (int kc = 0; kc < 2; ++kc)
#pragma unroll
    for (int dt = 0; dt < 2; ++dt)
      vf[kc][dt] = load_Bwt_n(vp + (size_t)(dt * 16) * 64 + kc * 32, 64, HD - dt * 16);

  int ncol = lane & 15, gsel = lane >> 4;

  for (int mt = 0; mt < 4; ++mt) {
    v16h qa = load_A_k(qp + mt * 16 * 192, 192, HD);
    v8f s0 = {}, s1 = {}, s2 = {}, s3 = {};
    s0 = wmma_f16(qa, kf[0], s0);
    s1 = wmma_f16(qa, kf[1], s1);
    s2 = wmma_f16(qa, kf[2], s2);
    s3 = wmma_f16(qa, kf[3], s3);
    float p[4][8];
#pragma unroll
    for (int r = 0; r < 8; ++r) { p[0][r] = s0[r]; p[1][r] = s1[r]; p[2][r] = s2[r]; p[3][r] = s3[r]; }

#pragma unroll
    for (int r = 0; r < 8; ++r) {
      int row = mt * 16 + r + (gsel << 3);
      int vr = mval(w, row);
      float mx = -1e30f;
#pragma unroll
      for (int c = 0; c < 4; ++c) {
        int col = c * 16 + ncol;
        float bl = btab[rpidx(row, col) * NHEAD + head];
        float mk = (mval(w, col) != vr) ? -100.0f : 0.0f;
        p[c][r] += bl + mk;
        mx = fmaxf(mx, p[c][r]);
      }
#pragma unroll
      for (int off = 8; off >= 1; off >>= 1) mx = fmaxf(mx, __shfl_xor(mx, off, 32));
      float sum = 0.f;
#pragma unroll
      for (int c = 0; c < 4; ++c) { p[c][r] = __expf(p[c][r] - mx); sum += p[c][r]; }
#pragma unroll
      for (int off = 8; off >= 1; off >>= 1) sum += __shfl_xor(sum, off, 32);
      float inv = 1.0f / sum;
#pragma unroll
      for (int c = 0; c < 4; ++c)
        lds[head][(r + (gsel << 3)) * 64 + c * 16 + ncol] = (_Float16)(p[c][r] * inv);
    }
    asm volatile("s_wait_dscnt 0" ::: "memory");   // wave-local store->load ordering

    v8f o0 = {}, o1 = {};
#pragma unroll
    for (int kc = 0; kc < 2; ++kc) {
      v16h af = load_A(&lds[head][kc * 32], 64);   // ds_load, row-major 16x64 block
      o0 = wmma_f16(af, vf[kc][0], o0);
      o1 = wmma_f16(af, vf[kc][1], o1);
    }
    asm volatile("s_wait_dscnt 0" ::: "memory");   // reads done before next mt overwrites

#pragma unroll
    for (int dt = 0; dt < 2; ++dt) {
      int d = dt * 16 + ncol;
      if (d < HD) {
        _Float16* op = ao16 + (size_t)win * NTOK * CC_ + head * HD + d;
        v8f& o = dt ? o1 : o0;
#pragma unroll
        for (int r = 0; r < 8; ++r) {
          int m = mt * 16 + r + (gsel << 3);
          op[(size_t)m * CC_] = (_Float16)o[r];
        }
      }
    }
  }
}

// ---------------------------------------------------------------------------
// k_proj: attnout(64x96) @ proj_w^T + b, window reverse + unshift + residual
// ---------------------------------------------------------------------------
__global__ void k_proj(const _Float16* __restrict__ ao16, const _Float16* __restrict__ wp,
                       const float* __restrict__ pb, const float* __restrict__ x,
                       float* __restrict__ xbuf) {
  int win = blockIdx.x;
  int wave = threadIdx.x >> 5, lane = threadIdx.x & 31;
  int b = win >> 6, w = win & 63;
  const _Float16* Ab = ao16 + (size_t)win * NTOK * CC_;
  int ncol = lane & 15, gsel = lane >> 4;
  for (int t = wave; t < 24; t += 4) {
    int mt = t / 6, nt = t % 6;
    v8f acc = {};
#pragma unroll
    for (int kt = 0; kt < 3; ++kt) {
      v16h a  = load_A(Ab + mt * 16 * CC_ + kt * 32, CC_);
      v16h bm = load_Bwt(wp + (nt * 16) * CC_ + kt * 32, CC_);
      acc = wmma_f16(a, bm, acc);
    }
    int col = nt * 16 + ncol;
    float bias = pb[col];
#pragma unroll
    for (int r = 0; r < 8; ++r) {
      int m = mt * 16 + r + (gsel << 3);
      int pr = (((w >> 3) << 3) + (m >> 3) + SSZ) & 63;
      int pc = (((w & 7) << 3) + (m & 7) + SSZ) & 63;
      size_t idx = (((size_t)b * LL_) + pr * 64 + pc) * CC_ + col;
      xbuf[idx] = x[idx] + acc[r] + bias;
    }
  }
}

// ---------------------------------------------------------------------------
// k_fc1: 64-row GEMM (K=96, N=384) + exact GELU. fc1_w panel (384x96 f16,
// 72KB) DMA'd into LDS by TDM, B-fragments then come from ds_load.
// ---------------------------------------------------------------------------
__global__ void k_fc1(const _Float16* __restrict__ h2n, const _Float16* __restrict__ w1,
                      const float* __restrict__ b1, _Float16* __restrict__ f1o) {
  extern __shared__ _Float16 wlds[];
  int rb = blockIdx.x;
  int wave = threadIdx.x >> 5, lane = threadIdx.x & 31;
  if (wave == 0) {
    tdm_load_2d(w1, 0u, 96u, 384u);                 // whole 384x96 panel -> LDS
    __builtin_amdgcn_s_wait_tensorcnt(0);
  }
  __syncthreads();
  const _Float16* Ab = h2n + (size_t)rb * 64 * CC_;
  int ncol = lane & 15, gsel = lane >> 4;
  for (int t = wave; t < 96; t += 8) {
    int mt = t / 24, nt = t % 24;
    v8f acc = {};
#pragma unroll
    for (int kt = 0; kt < 3; ++kt) {
      v16h a  = load_A(Ab + mt * 16 * CC_ + kt * 32, CC_);
      v16h bm = load_Bwt(wlds + (nt * 16) * CC_ + kt * 32, CC_);
      acc = wmma_f16(a, bm, acc);
    }
    int col = nt * 16 + ncol;
    float bias = b1[col];
    _Float16* op = f1o + (size_t)rb * 64 * 384 + col;
#pragma unroll
    for (int r = 0; r < 8; ++r) {
      int m = mt * 16 + r + (gsel << 3);
      float v = acc[r] + bias;
      float ge = 0.5f * v * (1.0f + erff(v * 0.70710678118654752f));
      op[(size_t)m * 384] = (_Float16)ge;
    }
  }
}

// ---------------------------------------------------------------------------
// k_fc2: 64-row GEMM (K=384, N=96) + bias + residual -> d_out. fc2_w panel
// (96x384 f16, 72KB) staged via TDM into LDS.
// ---------------------------------------------------------------------------
__global__ void k_fc2(const _Float16* __restrict__ f1o, const _Float16* __restrict__ w2,
                      const float* __restrict__ b2, const float* __restrict__ xbuf,
                      float* __restrict__ out) {
  extern __shared__ _Float16 wlds[];
  int rb = blockIdx.x;
  int wave = threadIdx.x >> 5, lane = threadIdx.x & 31;
  if (wave == 0) {
    tdm_load_2d(w2, 0u, 384u, 96u);                 // whole 96x384 panel -> LDS
    __builtin_amdgcn_s_wait_tensorcnt(0);
  }
  __syncthreads();
  const _Float16* Ab = f1o + (size_t)rb * 64 * 384;
  int ncol = lane & 15, gsel = lane >> 4;
  for (int t = wave; t < 24; t += 8) {
    int mt = t / 6, nt = t % 6;
    v8f acc = {};
#pragma unroll
    for (int kt = 0; kt < 12; ++kt) {
      if (kt < 11) __builtin_prefetch(Ab + mt * 16 * 384 + (kt + 1) * 32, 0, 1);
      v16h a  = load_A(Ab + mt * 16 * 384 + kt * 32, 384);
      v16h bm = load_Bwt(wlds + (nt * 16) * 384 + kt * 32, 384);
      acc = wmma_f16(a, bm, acc);
    }
    int col = nt * 16 + ncol;
    float bias = b2[col];
#pragma unroll
    for (int r = 0; r < 8; ++r) {
      int m = mt * 16 + r + (gsel << 3);
      size_t idx = ((size_t)rb * 64 + m) * CC_ + col;
      out[idx] = xbuf[idx] + acc[r] + bias;
    }
  }
}

// ---------------------------------------------------------------------------
extern "C" void kernel_launch(void* const* d_in, const int* in_sizes, int n_in,
                              void* d_out, int out_size, void* d_ws, size_t ws_size,
                              hipStream_t stream) {
  (void)in_sizes; (void)n_in; (void)out_size; (void)ws_size;
  const float* x      = (const float*)d_in[0];
  const float* n1g    = (const float*)d_in[1];
  const float* n1b    = (const float*)d_in[2];
  const float* qkv_w  = (const float*)d_in[3];
  const float* qkv_b  = (const float*)d_in[4];
  const float* btab   = (const float*)d_in[5];
  const float* proj_w = (const float*)d_in[6];
  const float* proj_b = (const float*)d_in[7];
  const float* n2g    = (const float*)d_in[8];
  const float* n2b    = (const float*)d_in[9];
  const float* fc1_w  = (const float*)d_in[10];
  const float* fc1_b  = (const float*)d_in[11];
  const float* fc2_w  = (const float*)d_in[12];
  const float* fc2_b  = (const float*)d_in[13];
  float* out = (float*)d_out;

  char* ws = (char*)d_ws;
  size_t off = 0;
  auto carve = [&](size_t bytes) -> void* {
    void* p = ws + off;
    off = (off + bytes + 255) & ~(size_t)255;
    return p;
  };
  const size_t TOK = (size_t)NWIN * NTOK;              // 262144 tokens
  _Float16* wq16  = (_Float16*)carve(288 * 96 * sizeof(_Float16));
  _Float16* wp16  = (_Float16*)carve(96 * 96 * sizeof(_Float16));
  _Float16* w116  = (_Float16*)carve(384 * 96 * sizeof(_Float16));
  _Float16* w216  = (_Float16*)carve(96 * 384 * sizeof(_Float16));
  _Float16* h16   = (_Float16*)carve(TOK * CC_ * sizeof(_Float16));
  _Float16* qk16  = (_Float16*)carve(TOK * 192 * sizeof(_Float16));
  _Float16* vT16  = (_Float16*)carve(TOK * 96 * sizeof(_Float16));
  _Float16* ao16  = (_Float16*)carve(TOK * CC_ * sizeof(_Float16));
  float*    xbuf  = (float*)   carve(TOK * CC_ * sizeof(float));
  _Float16* h2n16 = (_Float16*)carve(TOK * CC_ * sizeof(_Float16));
  _Float16* f1o16 = (_Float16*)carve(TOK * 384 * sizeof(_Float16));

  k_cvt<<<(288 * 96 + 255) / 256, 256, 0, stream>>>(qkv_w, wq16, 288 * 96);
  k_cvt<<<(96 * 96 + 255) / 256, 256, 0, stream>>>(proj_w, wp16, 96 * 96);
  k_cvt<<<(384 * 96 + 255) / 256, 256, 0, stream>>>(fc1_w, w116, 384 * 96);
  k_cvt<<<(96 * 384 + 255) / 256, 256, 0, stream>>>(fc2_w, w216, 96 * 384);

  k_ln1<<<TOK / 8, 256, 0, stream>>>(x, n1g, n1b, h16);
  k_qkv<<<NWIN, 256, 0, stream>>>(h16, wq16, qkv_b, qk16, vT16);
  k_attn<<<NWIN, 128, 0, stream>>>(qk16, vT16, btab, ao16);
  k_proj<<<NWIN, 128, 0, stream>>>(ao16, wp16, proj_b, x, xbuf);
  k_ln2<<<TOK / 8, 256, 0, stream>>>(xbuf, n2g, n2b, h2n16);
  k_fc1<<<NWIN, 256, 384 * 96 * sizeof(_Float16), stream>>>(h2n16, w116, fc1_b, f1o16);
  k_fc2<<<NWIN, 256, 96 * 384 * sizeof(_Float16), stream>>>(f1o16, w216, fc2_b, xbuf, out);
}
